// GRU_19524921328134
// MI455X (gfx1250) — compile-verified, matrix operand order checked by reading, not run on hardware
//
#include <hip/hip_runtime.h>

// GTrXL gating unit, fused, bf16 WMMA (v_wmma_f32_16x16x32_bf16) on gfx1250.
// T=256, B=64, D=512 -> M = 16384 rows.

typedef unsigned short ushort_t;
typedef __attribute__((ext_vector_type(16))) __bf16 v16bf;
typedef __attribute__((ext_vector_type(8)))  float  v8f;

#define M_TOTAL 16384
#define DDIM    512
#define KC      1024          // concat K (y | x)
#define MWG     64            // rows per workgroup
#define LSTRIDE 1040          // LDS row stride in bf16 elements (1024 + 16 pad)

__device__ __forceinline__ ushort_t f2bf(float f) {
    unsigned int u = __float_as_uint(f);
    unsigned int r = u + 0x7FFFu + ((u >> 16) & 1u);   // round-to-nearest-even
    return (ushort_t)(r >> 16);
}
__device__ __forceinline__ float bf2f(ushort_t h) {
    return __uint_as_float(((unsigned int)h) << 16);
}

union FragU { v16bf v; uint4 q[2]; };

// Load one 16x32 bf16 fragment. `p` points at (tile_row0, k_chunk_col0);
// per-lane: row = lane%16, k-chunks at kbase and kbase+16 (kbase = (lane/16)*8).
__device__ __forceinline__ v16bf load_frag(const ushort_t* p, int stride, int lane) {
    const int r  = lane & 15;
    const int kb = (lane >> 4) << 3;
    const ushort_t* q = p + (size_t)r * stride + kb;
    FragU f;
    f.q[0] = *(const uint4*)(q);
    f.q[1] = *(const uint4*)(q + 16);
    return f.v;
}

__device__ __forceinline__ v8f wmma_bf16(v16bf a, v16bf b, v8f c) {
    return __builtin_amdgcn_wmma_f32_16x16x32_bf16(
        /*neg_a=*/false, a, /*neg_b=*/false, b,
        /*c_mod=*/(short)0, c, /*reuse_a=*/false, /*reuse_b=*/false);
}

// ---------------- pack kernels: fp32 -> bf16 workspace ----------------

__global__ void pack_yx(const float* __restrict__ x, const float* __restrict__ y,
                        ushort_t* __restrict__ yx) {
    size_t i = (size_t)blockIdx.x * 256 + threadIdx.x;   // over M*1024
    size_t m = i >> 10;
    int    c = (int)(i & 1023);
    float v = (c < DDIM) ? y[m * DDIM + c] : x[m * DDIM + (c - DDIM)];
    yx[i] = f2bf(v);
}

__global__ void pack_wrz(const float* __restrict__ Wr, const float* __restrict__ Ur,
                         const float* __restrict__ Wz, const float* __restrict__ Uz,
                         ushort_t* __restrict__ Wrz) {
    int i = blockIdx.x * 256 + threadIdx.x;              // over 1024*1024
    int n = i >> 10, k = i & 1023;
    const float* a = (n < DDIM) ? ((k < DDIM) ? Wr : Ur)
                                : ((k < DDIM) ? Wz : Uz);
    Wrz[i] = f2bf(a[(size_t)(n & 511) * DDIM + (k & 511)]);
}

__global__ void pack_w(const float* __restrict__ W, ushort_t* __restrict__ o) {
    int i = blockIdx.x * 256 + threadIdx.x;              // over 512*512
    o[i] = f2bf(W[i]);
}

// ---------------- fused GTrXL gate kernel ----------------
// grid = M/MWG = 256 workgroups, 512 threads (16 waves).
// wave w: mg = w>>3 (2 M-tiles: rows mg*32 .. mg*32+31),
//         ng = w&7  (4 N-tiles: cols ng*64 .. ng*64+63).

extern __shared__ ushort_t smem[];   // MWG * LSTRIDE bf16 elements

__global__ void __launch_bounds__(512)
gtrxl_gate(const ushort_t* __restrict__ yx,   // M x 1024 bf16  [y|x]
           const ushort_t* __restrict__ Wrz,  // 1024 x 1024 bf16 [[Wr|Ur];[Wz|Uz]]
           const ushort_t* __restrict__ Wgb,  // 512 x 512 bf16
           const ushort_t* __restrict__ Ugb,  // 512 x 512 bf16
           const float*    __restrict__ x32,  // M x 512 fp32 (original x)
           const float*    __restrict__ bg,   // 512
           float*          __restrict__ out)  // M x 512 fp32
{
    const int tid  = threadIdx.x;
    const int lane = tid & 31;
    const int w    = tid >> 5;
    const int mg   = w >> 3;      // 0..1
    const int ng   = w & 7;       // 0..7
    const int rowbase = blockIdx.x * MWG;

    // ---- stage 64 rows of [y|x] (bf16) into LDS, 16B chunks ----
    for (int c = tid; c < MWG * (KC / 8); c += 512) {
        int r  = c >> 7;          // row (1024/8 = 128 chunks per row)
        int cc = c & 127;
        uint4 v = *(const uint4*)(yx + (size_t)(rowbase + r) * KC + cc * 8);
        *((uint4*)(smem + r * LSTRIDE + cc * 8)) = v;
    }
    __syncthreads();

    const v8f vzero = {0.f, 0.f, 0.f, 0.f, 0.f, 0.f, 0.f, 0.f};
    v8f accR[2][4], accZ[2][4], accH[2][4];
    for (int mi = 0; mi < 2; ++mi)
        for (int j = 0; j < 4; ++j) { accR[mi][j] = vzero; accZ[mi][j] = vzero; accH[mi][j] = vzero; }

    // ---- GEMM phase 1a: kc in [0,16) -> r_pre, z_pre, ywg (branch-free) ----
    for (int kc = 0; kc < DDIM / 32; ++kc) {
        v16bf a0 = load_frag(smem + (mg * 32 +  0) * LSTRIDE + kc * 32, LSTRIDE, lane);
        v16bf a1 = load_frag(smem + (mg * 32 + 16) * LSTRIDE + kc * 32, LSTRIDE, lane);
        for (int j = 0; j < 4; ++j) {
            const int n = (ng * 4 + j) * 16;
            const ushort_t* pr = Wrz + (size_t)n * KC + kc * 32;
            __builtin_prefetch(pr + 32, 0, 1);
            // Load all three B fragments first -> one load clause, partial waits.
            v16bf br = load_frag(pr, KC, lane);
            v16bf bz = load_frag(Wrz + (size_t)(n + DDIM) * KC + kc * 32, KC, lane);
            v16bf bw = load_frag(Wgb + (size_t)n * DDIM + kc * 32, DDIM, lane);
            accR[0][j] = wmma_bf16(a0, br, accR[0][j]);
            accR[1][j] = wmma_bf16(a1, br, accR[1][j]);
            accZ[0][j] = wmma_bf16(a0, bz, accZ[0][j]);
            accZ[1][j] = wmma_bf16(a1, bz, accZ[1][j]);
            accH[0][j] = wmma_bf16(a0, bw, accH[0][j]);
            accH[1][j] = wmma_bf16(a1, bw, accH[1][j]);
        }
    }
    // ---- GEMM phase 1b: kc in [16,32) -> r_pre, z_pre only (x-half of K) ----
    for (int kc = DDIM / 32; kc < KC / 32; ++kc) {
        v16bf a0 = load_frag(smem + (mg * 32 +  0) * LSTRIDE + kc * 32, LSTRIDE, lane);
        v16bf a1 = load_frag(smem + (mg * 32 + 16) * LSTRIDE + kc * 32, LSTRIDE, lane);
        for (int j = 0; j < 4; ++j) {
            const int n = (ng * 4 + j) * 16;
            const ushort_t* pr = Wrz + (size_t)n * KC + kc * 32;
            if (kc + 1 < KC / 32) __builtin_prefetch(pr + 32, 0, 1);
            v16bf br = load_frag(pr, KC, lane);
            v16bf bz = load_frag(Wrz + (size_t)(n + DDIM) * KC + kc * 32, KC, lane);
            accR[0][j] = wmma_bf16(a0, br, accR[0][j]);
            accR[1][j] = wmma_bf16(a1, br, accR[1][j]);
            accZ[0][j] = wmma_bf16(a0, bz, accZ[0][j]);
            accZ[1][j] = wmma_bf16(a1, bz, accZ[1][j]);
        }
    }

    // ---- r = sigmoid(r_pre); overwrite x-half of LDS with rx = r*x (bf16) ----
    __syncthreads();   // all phase-1 LDS reads done before overwrite
    for (int mi = 0; mi < 2; ++mi) {
        const int rb = mg * 32 + mi * 16 + ((lane >> 4) << 3);
        for (int j = 0; j < 4; ++j) {
            const int n0 = (ng * 4 + j) * 16 + (lane & 15);  // d-index 0..511
            for (int v = 0; v < 8; ++v) {
                const int row = rb + v;
                float r  = 1.f / (1.f + __expf(-accR[mi][j][v]));
                const int off = row * LSTRIDE + DDIM + n0;
                float xv = bf2f(smem[off]);
                smem[off] = f2bf(r * xv);
            }
        }
    }
    __syncthreads();

    // ---- GEMM phase 2: h_pre = ywg + rx @ Ug^T  (K=512, A = rx from LDS) ----
    for (int kc = 0; kc < DDIM / 32; ++kc) {
        v16bf a0 = load_frag(smem + (mg * 32 +  0) * LSTRIDE + DDIM + kc * 32, LSTRIDE, lane);
        v16bf a1 = load_frag(smem + (mg * 32 + 16) * LSTRIDE + DDIM + kc * 32, LSTRIDE, lane);
        for (int j = 0; j < 4; ++j) {
            const int n = (ng * 4 + j) * 16;
            const ushort_t* pu = Ugb + (size_t)n * DDIM + kc * 32;
            if (kc + 1 < DDIM / 32) __builtin_prefetch(pu + 32, 0, 1);
            v16bf bu = load_frag(pu, DDIM, lane);
            accH[0][j] = wmma_bf16(a0, bu, accH[0][j]);
            accH[1][j] = wmma_bf16(a1, bu, accH[1][j]);
        }
    }

    // ---- epilogue: z = sigmoid(z_pre - bg); h = tanh; g = (1-z)x + z h ----
    for (int j = 0; j < 4; ++j) {
        const int e = (ng * 4 + j) * 16 + (lane & 15);
        const float b = bg[e];
        for (int mi = 0; mi < 2; ++mi) {
            const int rb = mg * 32 + mi * 16 + ((lane >> 4) << 3);
            for (int v = 0; v < 8; ++v) {
                const int m = rowbase + rb + v;
                float z = 1.f / (1.f + __expf(-(accZ[mi][j][v] - b)));
                float h = tanhf(accH[mi][j][v]);
                float xv = x32[(size_t)m * DDIM + e];
                out[(size_t)m * DDIM + e] = (1.f - z) * xv + z * h;
            }
        }
    }
}

// ---------------- host side ----------------

extern "C" void kernel_launch(void* const* d_in, const int* in_sizes, int n_in,
                              void* d_out, int out_size, void* d_ws, size_t ws_size,
                              hipStream_t stream) {
    const float* x  = (const float*)d_in[0];
    const float* y  = (const float*)d_in[1];
    const float* Wr = (const float*)d_in[2];
    const float* Ur = (const float*)d_in[3];
    const float* Wz = (const float*)d_in[4];
    const float* Uz = (const float*)d_in[5];
    const float* Wg = (const float*)d_in[6];
    const float* Ug = (const float*)d_in[7];
    const float* bg = (const float*)d_in[8];

    ushort_t* ws  = (ushort_t*)d_ws;
    ushort_t* yx  = ws;                                   // 16384*1024 bf16
    ushort_t* wrz = yx  + (size_t)M_TOTAL * KC;           // 1024*1024
    ushort_t* wgb = wrz + (size_t)KC * KC;                // 512*512
    ushort_t* ugb = wgb + (size_t)DDIM * DDIM;            // 512*512

    pack_yx <<<(M_TOTAL * KC) / 256, 256, 0, stream>>>(x, y, yx);
    pack_wrz<<<(KC * KC) / 256,      256, 0, stream>>>(Wr, Ur, Wz, Uz, wrz);
    pack_w  <<<(DDIM * DDIM) / 256,  256, 0, stream>>>(Wg, wgb);
    pack_w  <<<(DDIM * DDIM) / 256,  256, 0, stream>>>(Ug, ugb);

    const size_t lds_bytes = (size_t)MWG * LSTRIDE * sizeof(ushort_t); // ~130 KB (<=320 KB/WGP)
    gtrxl_gate<<<M_TOTAL / MWG, 512, lds_bytes, stream>>>(
        yx, wrz, wgb, ugb, x, bg, (float*)d_out);
}
